// EncoderLayer_59751585022190
// MI455X (gfx1250) — matrix operand order repaired and checked by reading, not compiled, not run on hardware
//
#include <hip/hip_runtime.h>
#include <hip/hip_bf16.h>

typedef __bf16 bf16;
typedef __attribute__((ext_vector_type(16))) __bf16 v16bf;
typedef __attribute__((ext_vector_type(8)))  float  v8f;
typedef unsigned u32;
typedef __attribute__((ext_vector_type(4))) u32 u32x4;
typedef __attribute__((ext_vector_type(8))) u32 u32x8;

union AF16 { v16bf v; uint4 u[2]; };

constexpr int cL = 2048, cN = 2, cE = 1024, cH = 16, cD = 64, cFFN = 4096;
constexpr int cM = cL * cN;            // 4096 token rows
constexpr float cSCALE = 0.125f;       // D^-0.5

static __device__ __forceinline__ v8f wmma_bf16(const v16bf& a, const v16bf& b, v8f c) {
  return __builtin_amdgcn_wmma_f32_16x16x32_bf16(false, a, false, b, (short)0, c, false, false);
}

// LDS byte offset (wave-relative) of a __shared__ object: generic -> AS(3) cast.
static __device__ __forceinline__ unsigned lds_off(void* p) {
  return (unsigned)(unsigned long long)(__attribute__((address_space(3))) void*)p;
}

// TDM: load a 2D tile (tile_d0 x tile_d1, bf16) from a row-major tensor into LDS.
// D# group0/group1 built per CDNA5 ISA 8.3/8.4; 2D -> groups 2/3 NULL.
static __device__ __forceinline__ void tdm_load_2d(
    unsigned ldsaddr, const void* gaddr, u32 tensor_d0, u32 tensor_d1,
    u32 tile_d0, u32 tile_d1, u32 stride0_elems) {
  const unsigned long long ga = (unsigned long long)gaddr;
  u32x4 g0;
  g0[0] = 1u;                                        // count=1, user descriptor
  g0[1] = ldsaddr;                                   // lds_addr (bytes)
  g0[2] = (u32)ga;                                   // global_addr[31:0]
  g0[3] = ((u32)(ga >> 32) & 0x1FFFFFFu) | (2u << 30); // global_addr[56:32], type=2
  u32x8 g1;
  g1[0] = 1u << 16;                                  // data_size=1 (2 bytes)
  g1[1] = (tensor_d0 & 0xFFFFu) << 16;               // tensor_dim0[15:0]
  g1[2] = (tensor_d0 >> 16) | ((tensor_d1 & 0xFFFFu) << 16); // dim0 hi | dim1 lo
  g1[3] = (tensor_d1 >> 16) | (tile_d0 << 16);       // dim1 hi | tile_dim0
  g1[4] = tile_d1;                                   // tile_dim1 (tile_dim2=0)
  g1[5] = stride0_elems;                             // tensor_dim0_stride[31:0]
  g1[6] = 0;                                         // stride0 hi / stride1 lo
  g1[7] = 0;
  asm volatile("tensor_load_to_lds %0, %1" :: "s"(g0), "s"(g1) : "memory");
}

// ---------------------------------------------------------------- utilities
__global__ void f2bf_kernel(const float* __restrict__ in, bf16* __restrict__ out, int n) {
  int i = blockIdx.x * 256 + threadIdx.x;
  if (i < n) out[i] = (bf16)in[i];
}

__global__ __launch_bounds__(256) void layernorm_kernel(
    const float* __restrict__ x, const float* __restrict__ w,
    const float* __restrict__ b, bf16* __restrict__ out) {
  __shared__ float red[16];
  const int m = blockIdx.x, t = threadIdx.x;
  const float* xr = x + (long)m * cE;
  float v[4], s = 0.f, ss = 0.f;
  for (int i = 0; i < 4; ++i) { v[i] = xr[t + i * 256]; s += v[i]; ss += v[i] * v[i]; }
  for (int d = 1; d < 32; d <<= 1) { s += __shfl_xor(s, d, 32); ss += __shfl_xor(ss, d, 32); }
  if ((t & 31) == 0) { red[t >> 5] = s; red[8 + (t >> 5)] = ss; }
  __syncthreads();
  float S = 0.f, SS = 0.f;
  for (int i = 0; i < 8; ++i) { S += red[i]; SS += red[8 + i]; }
  const float mu  = S * (1.f / cE);
  const float var = SS * (1.f / cE) - mu * mu;
  const float inv = rsqrtf(var + 1e-6f);
  for (int i = 0; i < 4; ++i) {
    int e = t + i * 256;
    out[(long)m * cE + e] = (bf16)((v[i] - mu) * inv * w[e] + b[e]);
  }
}

// split interleaved qkv (M x 3E fp32) -> Q*scale, K as (nh,L,D) bf16; V^T as (nh,D,L) bf16
__global__ void split_qkv_kernel(const float* __restrict__ qkv,
                                 bf16* __restrict__ Q, bf16* __restrict__ K,
                                 bf16* __restrict__ Vt) {
  long i = (long)blockIdx.x * 256 + threadIdx.x;   // over M*E
  if (i >= (long)cM * cE) return;
  const int m = (int)(i >> 10);          // /E
  const int e = (int)(i & (cE - 1));
  const int h = e >> 6, d = e & 63;
  const int l = m >> 1, n = m & 1;       // row m = l*N + n
  const float* p = qkv + (long)m * (3 * cE) + (long)e * 3;
  const long nh = (long)n * cH + h;
  Q [(nh * cL + l) * cD + d] = (bf16)(p[0] * cSCALE);
  K [(nh * cL + l) * cD + d] = (bf16)(p[1]);
  Vt[(nh * cD + d) * cL + l] = (bf16)(p[2]);
}

// ---------------------------------------------------------------- WMMA GEMM
// C[m,n] = sum_k A[m,k] * W[n,k] (+bias, optional residual / GELU)
// 128x128 block tile, 8 waves (4 row-groups x 2 col-groups), 32x64 per wave.
// A tile staged by TDM (tensor_load_to_lds, TENSORcnt), B tile by per-lane
// GLOBAL_LOAD_ASYNC_TO_LDS_B128 (ASYNCcnt); both double-buffered.
#define BM 128
#define BN 128
#define BK 32

__global__ __launch_bounds__(256, 1) void gemm_bf16_kernel(
    const bf16* __restrict__ A, const bf16* __restrict__ W,
    const float* __restrict__ bias, const float* __restrict__ res,
    float* __restrict__ outF, bf16* __restrict__ outB, int gelu,
    int M, int Nout, int K) {
  __shared__ bf16 sA[2][BM * BK];
  __shared__ bf16 sB[2][BN * BK];
  const int tid  = threadIdx.x;
  const int lane = tid & 31;
  const int wave = tid >> 5;            // 0..7
  const int lh   = lane >> 4;           // lane half
  const int l16  = lane & 15;
  const int wrow = (wave >> 1) * 32;    // 0,32,64,96
  const int wcol = (wave & 1) * 64;     // 0,64
  const long bm = (long)blockIdx.x * BM;
  const long bn = (long)blockIdx.y * BN;

  v8f acc[2][4];
  for (int r = 0; r < 2; ++r)
    for (int c = 0; c < 4; ++c)
      for (int i = 0; i < 8; ++i) acc[r][c][i] = 0.f;

  // B staging: each of 256 threads moves 32B per tile
  const int sr = tid >> 1;              // 0..127 (tile row)
  const int sc = (tid & 1) * 16;        // 0 or 16 (tile col)
  const bf16* gB = W + (bn + sr) * (long)K + sc;
  const unsigned lb[2] = { lds_off(&sB[0][sr * BK + sc]), lds_off(&sB[1][sr * BK + sc]) };
  // A staging: one TDM descriptor per tile (issued by wave 0)
  const bf16* gA = A + bm * (long)K;
  const unsigned laT[2] = { lds_off(&sA[0][0]), lds_off(&sA[1][0]) };

  auto stage = [&](int buf, long k0) {
    const unsigned long long b = (unsigned long long)(gB + k0);
    asm volatile(
        "global_load_async_to_lds_b128 %0, %1, off\n\t"
        "global_load_async_to_lds_b128 %0, %1, off offset:16"
        :: "v"(lb[buf]), "v"(b) : "memory");
    if (wave == 0)
      tdm_load_2d(laT[buf], gA + k0, (u32)K, (u32)BM, BK, BM, (u32)K);
  };

  stage(0, 0);
  int cur = 0;
  for (long k0 = 0; k0 < K; k0 += BK) {
    asm volatile("s_wait_asynccnt 0" ::: "memory");     // B tile resident
    if (wave == 0) __builtin_amdgcn_s_wait_tensorcnt(0); // A tile resident
    __syncthreads();
    if (k0 + BK < K) stage(cur ^ 1, k0 + BK);           // overlap next copy

    AF16 a0, a1;
    {
      const bf16* p = &sA[cur][(wrow + l16) * BK + lh * 8];
      a0.u[0] = *(const uint4*)p;
      a0.u[1] = *(const uint4*)(p + 16);
      p += 16 * BK;
      a1.u[0] = *(const uint4*)p;
      a1.u[1] = *(const uint4*)(p + 16);
    }
    for (int c = 0; c < 4; ++c) {
      AF16 b;
      const bf16* p = &sB[cur][(wcol + c * 16 + l16) * BK + lh * 8];
      b.u[0] = *(const uint4*)p;
      b.u[1] = *(const uint4*)(p + 16);
      acc[0][c] = wmma_bf16(a0.v, b.v, acc[0][c]);
      acc[1][c] = wmma_bf16(a1.v, b.v, acc[1][c]);
    }
    __syncthreads();
    cur ^= 1;
  }

  // epilogue: low register pressure — one (r,c) tile at a time, mode branch
  // hoisted, single pointer advanced by Nout per row.
  for (int r = 0; r < 2; ++r)
    for (int c = 0; c < 4; ++c) {
      const long col  = bn + wcol + c * 16 + l16;
      const long row0 = bm + wrow + r * 16 + lh * 8;
      const float bv  = bias[col];
      if (gelu) {
        bf16* po = outB + row0 * Nout + col;
        #pragma unroll 1
        for (int i = 0; i < 8; ++i) {
          float v = acc[r][c][i] + bv;
          v = 0.5f * v * (1.f + erff(v * 0.70710678118654752f));
          *po = (bf16)v;
          po += Nout;
        }
      } else if (res) {
        const float* pr = res + row0 * Nout + col;
        float* po = outF + row0 * Nout + col;
        #pragma unroll 1
        for (int i = 0; i < 8; ++i) {
          *po = acc[r][c][i] + bv + *pr;
          pr += Nout; po += Nout;
        }
      } else {
        float* po = outF + row0 * Nout + col;
        #pragma unroll 1
        for (int i = 0; i < 8; ++i) {
          *po = acc[r][c][i] + bv;
          po += Nout;
        }
      }
    }
}

// ---------------------------------------------------------------- flash attention
// one wave handles 16 query rows of one (n,h); streams 32 keys/iter
__global__ __launch_bounds__(32) void attn_kernel(
    const bf16* __restrict__ Qm, const bf16* __restrict__ Km,
    const bf16* __restrict__ Vtm, bf16* __restrict__ Ob) {
  __shared__ bf16 sP[16 * 32];
  const int lane = threadIdx.x;
  const int lh = lane >> 4, l16 = lane & 15;
  const int nh = blockIdx.y;           // n*H + h
  const int qt = blockIdx.x;           // 16-row query tile
  const bf16* Qb = Qm  + ((long)nh * cL + qt * 16) * cD;
  const bf16* Kb = Km  + (long)nh * cL * cD;
  const bf16* Vb = Vtm + (long)nh * cD * cL;

  AF16 qf[2];                          // Q tile 16x64 = two A-frags (k=0..31, 32..63)
  for (int s = 0; s < 2; ++s) {
    const bf16* p = Qb + l16 * cD + s * 32 + lh * 16;
    qf[s].u[0] = *(const uint4*)p;
    qf[s].u[1] = *(const uint4*)(p + 8);
  }
  float mrun[8], lrun[8];
  v8f o[4];
  for (int i = 0; i < 8; ++i) { mrun[i] = -1e30f; lrun[i] = 0.f; }
  for (int c = 0; c < 4; ++c) for (int i = 0; i < 8; ++i) o[c][i] = 0.f;

  for (int jb = 0; jb < cL; jb += 32) {
    if (jb + 32 < cL) {
      __builtin_prefetch(Kb + (long)(jb + 32 + l16) * cD, 0, 3);
      __builtin_prefetch(Vb + (long)l16 * cL + jb + 32, 0, 3);
    }
    v8f s0, s1;
    for (int i = 0; i < 8; ++i) { s0[i] = 0.f; s1[i] = 0.f; }
    for (int s = 0; s < 2; ++s) {      // S = Q K^T (scale folded into Q)
      AF16 b;
      const bf16* p = Kb + (long)(jb + l16) * cD + s * 32 + lh * 16;
      b.u[0] = ((const uint4*)p)[0]; b.u[1] = ((const uint4*)p)[1];
      s0 = wmma_bf16(qf[s].v, b.v, s0);
      p = Kb + (long)(jb + 16 + l16) * cD + s * 32 + lh * 16;
      b.u[0] = ((const uint4*)p)[0]; b.u[1] = ((const uint4*)p)[1];
      s1 = wmma_bf16(qf[s].v, b.v, s1);
    }
    float alpha[8];
    for (int i = 0; i < 8; ++i) {      // online softmax, row = i + 8*lh
      float mx = fmaxf(s0[i], s1[i]);
      for (int d = 1; d < 16; d <<= 1) mx = fmaxf(mx, __shfl_xor(mx, d, 16));
      const float nm = fmaxf(mrun[i], mx);
      alpha[i] = __expf(mrun[i] - nm);
      mrun[i] = nm;
      const float p0 = __expf(s0[i] - nm), p1 = __expf(s1[i] - nm);
      float rs = p0 + p1;
      for (int d = 1; d < 16; d <<= 1) rs += __shfl_xor(rs, d, 16);
      lrun[i] = lrun[i] * alpha[i] + rs;
      const int row = i + lh * 8;
      sP[row * 32 + l16]      = (bf16)p0;
      sP[row * 32 + 16 + l16] = (bf16)p1;
    }
    __syncthreads();
    AF16 pf;                           // P as A-frag (16x32)
    {
      const bf16* p = sP + l16 * 32 + lh * 16;
      pf.u[0] = *(const uint4*)p;
      pf.u[1] = *(const uint4*)(p + 8);
    }
    for (int c = 0; c < 4; ++c) {      // O = diag(alpha) O + P V
      for (int i = 0; i < 8; ++i) o[c][i] *= alpha[i];
      AF16 b;
      const bf16* p = Vb + (long)(c * 16 + l16) * cL + jb + lh * 16;
      b.u[0] = ((const uint4*)p)[0]; b.u[1] = ((const uint4*)p)[1];
      o[c] = wmma_bf16(pf.v, b.v, o[c]);
    }
    __syncthreads();
  }
  const int n = nh >> 4, h = nh & 15;
  for (int c = 0; c < 4; ++c)
    for (int i = 0; i < 8; ++i) {
      const int  l   = qt * 16 + lh * 8 + i;
      const long row = (long)l * cN + n;
      const long col = (long)h * cD + c * 16 + l16;
      Ob[row * cE + col] = (bf16)(o[c][i] / lrun[i]);
    }
}

// ---------------------------------------------------------------- driver
extern "C" void kernel_launch(void* const* d_in, const int* in_sizes, int n_in,
                              void* d_out, int out_size, void* d_ws, size_t ws_size,
                              hipStream_t stream) {
  (void)in_sizes; (void)n_in; (void)out_size; (void)ws_size;
  const float* x        = (const float*)d_in[0];
  const float* qkv_proj = (const float*)d_in[1];
  const float* qkv_bias = (const float*)d_in[2];
  const float* out_proj = (const float*)d_in[3];
  const float* out_bias = (const float*)d_in[4];
  const float* ln1_w    = (const float*)d_in[5];
  const float* ln1_b    = (const float*)d_in[6];
  const float* proj1    = (const float*)d_in[7];
  const float* proj1_b  = (const float*)d_in[8];
  const float* proj2    = (const float*)d_in[9];
  const float* proj2_b  = (const float*)d_in[10];
  const float* ln2_w    = (const float*)d_in[11];
  const float* ln2_b    = (const float*)d_in[12];
  float* out = (float*)d_out;

  size_t off = 0;
  auto alloc = [&](size_t elems, size_t esz) -> void* {
    void* p = (char*)d_ws + off;
    off += (elems * esz + 255) & ~(size_t)255;
    return p;
  };
  bf16*  wqkv  = (bf16*) alloc((size_t)3 * cE * cE, 2);
  bf16*  wout  = (bf16*) alloc((size_t)cE * cE, 2);
  bf16*  w1    = (bf16*) alloc((size_t)cFFN * cE, 2);
  bf16*  w2    = (bf16*) alloc((size_t)cE * cFFN, 2);
  bf16*  hln   = (bf16*) alloc((size_t)cM * cE, 2);
  float* qkvf  = (float*)alloc((size_t)cM * 3 * cE, 4);
  bf16*  Q     = (bf16*) alloc((size_t)cN * cH * cL * cD, 2);
  bf16*  K     = (bf16*) alloc((size_t)cN * cH * cL * cD, 2);
  bf16*  Vt    = (bf16*) alloc((size_t)cN * cH * cD * cL, 2);
  bf16*  attnb = (bf16*) alloc((size_t)cM * cE, 2);
  float* x1    = (float*)alloc((size_t)cM * cE, 4);
  bf16*  h2    = (bf16*) alloc((size_t)cM * cE, 2);
  bf16*  hmid  = (bf16*) alloc((size_t)cM * cFFN, 2);

  // 1) weights fp32 -> bf16
  auto conv = [&](const float* src, bf16* dst, int n) {
    f2bf_kernel<<<(n + 255) / 256, 256, 0, stream>>>(src, dst, n);
  };
  conv(qkv_proj, wqkv, 3 * cE * cE);
  conv(out_proj, wout, cE * cE);
  conv(proj1,    w1,   cFFN * cE);
  conv(proj2,    w2,   cE * cFFN);

  // 2) LN1
  layernorm_kernel<<<cM, 256, 0, stream>>>(x, ln1_w, ln1_b, hln);
  // 3) QKV projection (M x 3E)
  gemm_bf16_kernel<<<dim3(cM / BM, 3 * cE / BN), 256, 0, stream>>>(
      hln, wqkv, qkv_bias, nullptr, qkvf, nullptr, 0, cM, 3 * cE, cE);
  // 4) split + layout for attention
  split_qkv_kernel<<<(cM * cE + 255) / 256, 256, 0, stream>>>(qkvf, Q, K, Vt);
  // 5) flash attention
  attn_kernel<<<dim3(cL / 16, cN * cH), 32, 0, stream>>>(Q, K, Vt, attnb);
  // 6) out projection + residual -> x1
  gemm_bf16_kernel<<<dim3(cM / BM, cE / BN), 256, 0, stream>>>(
      attnb, wout, out_bias, x, x1, nullptr, 0, cM, cE, cE);
  // 7) LN2
  layernorm_kernel<<<cM, 256, 0, stream>>>(x1, ln2_w, ln2_b, h2);
  // 8) FFN up + GELU (bf16 out)
  gemm_bf16_kernel<<<dim3(cM / BM, cFFN / BN), 256, 0, stream>>>(
      h2, w1, proj1_b, nullptr, nullptr, hmid, 1, cM, cFFN, cE);
  // 9) FFN down + residual -> d_out
  gemm_bf16_kernel<<<dim3(cM / BM, cE / BN), 256, 0, stream>>>(
      hmid, w2, proj2_b, x1, out, nullptr, 0, cM, cE, cFFN);
}
